// SAGEConv_14886356648743
// MI455X (gfx1250) — matrix operand order repaired and checked by reading, not compiled, not run on hardware
//
#include <hip/hip_runtime.h>

#define N_NODES 10000
#define N_EDGES 320000
#define D_IN    256
#define D_OUT   256
#define D_CAT   512   // 2 * D_IN

typedef __bf16 bf16_t;
typedef __bf16 v8bf   __attribute__((ext_vector_type(8)));
typedef __bf16 v16bf  __attribute__((ext_vector_type(16)));
typedef float  v8f    __attribute__((ext_vector_type(8)));

// ---------------------------------------------------------------------------
// 1) zero the accumulation buffers (summed [N,256] + deg [N])
// ---------------------------------------------------------------------------
__global__ void zero_kernel(float* __restrict__ p, unsigned n) {
    unsigned i = blockIdx.x * blockDim.x + threadIdx.x;
    if (i < n) p[i] = 0.0f;
}

// ---------------------------------------------------------------------------
// 2) edge scatter: summed[dst] += x[src]; deg[dst] += 1
//    one thread per (edge, float4-chunk); x rows are L2-resident (10 MB)
// ---------------------------------------------------------------------------
__global__ void scatter_kernel(const float* __restrict__ x,
                               const long long* __restrict__ ei,
                               float* __restrict__ summed,
                               float* __restrict__ deg) {
    unsigned tid = blockIdx.x * blockDim.x + threadIdx.x;   // e*64 + q
    if (tid >= (unsigned)N_EDGES * 64u) return;
    unsigned e = tid >> 6;
    unsigned q = tid & 63u;
    int src = (int)ei[e];
    int dst = (int)ei[(size_t)N_EDGES + e];
    const float4* xr = (const float4*)(x + (size_t)src * D_IN);
    float4 v = xr[q];
    float* s = summed + (size_t)dst * D_IN + q * 4u;
    atomicAdd(s + 0, v.x);
    atomicAdd(s + 1, v.y);
    atomicAdd(s + 2, v.z);
    atomicAdd(s + 3, v.w);
    if (q == 0) atomicAdd(deg + dst, 1.0f);
}

// ---------------------------------------------------------------------------
// 3) build bf16 concat input: inb[row][0:256]=x, inb[row][256:512]=summed/deg
// ---------------------------------------------------------------------------
__global__ void convert_kernel(const float* __restrict__ x,
                               const float* __restrict__ summed,
                               const float* __restrict__ deg,
                               bf16_t* __restrict__ inb) {
    unsigned tid = blockIdx.x * blockDim.x + threadIdx.x;   // N_NODES * 512
    if (tid >= (unsigned)N_NODES * D_CAT) return;
    unsigned row = tid >> 9;
    unsigned col = tid & 511u;
    float v;
    if (col < D_IN) {
        v = x[(size_t)row * D_IN + col];
    } else {
        v = summed[(size_t)row * D_IN + (col - D_IN)] / fmaxf(deg[row], 1.0f);
    }
    inb[tid] = (bf16_t)v;
}

// ---------------------------------------------------------------------------
// 4) transpose+convert weight:  Bt[j][k] = W[k][j]  (bf16, [256][512])
// ---------------------------------------------------------------------------
__global__ void wtrans_kernel(const float* __restrict__ w,
                              bf16_t* __restrict__ bt) {
    unsigned tid = blockIdx.x * blockDim.x + threadIdx.x;   // 256*512
    if (tid >= (unsigned)D_OUT * D_CAT) return;
    unsigned j = tid >> 9;       // output column
    unsigned k = tid & 511u;     // k index
    bt[tid] = (bf16_t)w[(size_t)k * D_OUT + j];
}

// ---------------------------------------------------------------------------
// 5) WMMA GEMM: raw[10000][256] = inb[10000][512] @ Bt^T
//    one wave32 per 16x16 output tile; K loop of v_wmma_f32_16x16x32_bf16
//
//    CDNA5 A fragment (16x32 bf16, ISA 7.12.2): lane half selects K offset
//    {0,8} for VGPR0-3 and {16,24} for VGPR4-7 -> two contiguous 16B loads at
//    row + ks + 8*half and row + ks + 16 + 8*half.
//    B fragment (32x16 bf16): same K order as A -> two contiguous 16B loads
//    at btrow + ks + 8*half and btrow + ks + 16 + 8*half.
// ---------------------------------------------------------------------------
__global__ void gemm_kernel(const bf16_t* __restrict__ A,
                            const bf16_t* __restrict__ Bt,
                            float* __restrict__ raw) {
    unsigned wave = blockIdx.x * (blockDim.x >> 5) + (threadIdx.x >> 5);
    unsigned lane = threadIdx.x & 31u;
    unsigned tm   = wave >> 4;     // 0..624  (row tile)
    unsigned tn   = wave & 15u;    // 0..15   (col tile)
    unsigned half = lane >> 4;     // 0 or 1
    unsigned l16  = lane & 15u;

    const bf16_t* arow = A  + (size_t)(tm * 16u + l16) * D_CAT;
    const bf16_t* brow = Bt + (size_t)(tn * 16u + l16) * D_CAT;

    v8f c = {};
    #pragma unroll 4
    for (unsigned ks = 0; ks < D_CAT; ks += 32u) {
        v8bf a0 = *(const v8bf*)(arow + ks + 8u * half);
        v8bf a1 = *(const v8bf*)(arow + ks + 16u + 8u * half);
        v8bf b0 = *(const v8bf*)(brow + ks + 8u * half);
        v8bf b1 = *(const v8bf*)(brow + ks + 16u + 8u * half);
        v16bf a = __builtin_shufflevector(a0, a1, 0, 1, 2, 3, 4, 5, 6, 7,
                                                  8, 9, 10, 11, 12, 13, 14, 15);
        v16bf b = __builtin_shufflevector(b0, b1, 0, 1, 2, 3, 4, 5, 6, 7,
                                                  8, 9, 10, 11, 12, 13, 14, 15);
        c = __builtin_amdgcn_wmma_f32_16x16x32_bf16(
                /*neg_a=*/false, a, /*neg_b=*/false, b,
                /*c_mod=*/(short)0, c, /*reuse_a=*/false, /*reuse_b=*/false);
    }

    // C/D layout: VGPR r -> M = r + 8*half, N = lane%16
    float* out0 = raw + (size_t)(tm * 16u) * D_OUT + tn * 16u + l16;
    #pragma unroll
    for (int r = 0; r < 8; ++r) {
        out0[(size_t)(r + 8u * half) * D_OUT] = c[r];
    }
}

// ---------------------------------------------------------------------------
// 6) per-row L2 normalize + bias; one 256-thread block per node row
// ---------------------------------------------------------------------------
__global__ void finalize_kernel(const float* __restrict__ raw,
                                const float* __restrict__ bias,
                                float* __restrict__ out) {
    __shared__ float red[256];
    unsigned row = blockIdx.x;
    unsigned t   = threadIdx.x;
    float v = raw[(size_t)row * D_OUT + t];
    red[t] = v * v;
    __syncthreads();
    for (unsigned s = 128; s > 0; s >>= 1) {
        if (t < s) red[t] += red[t + s];
        __syncthreads();
    }
    float norm = sqrtf(red[0]);
    out[(size_t)row * D_OUT + t] = v / norm + bias[t];
}

// ---------------------------------------------------------------------------
// launcher
// ---------------------------------------------------------------------------
extern "C" void kernel_launch(void* const* d_in, const int* in_sizes, int n_in,
                              void* d_out, int out_size, void* d_ws, size_t ws_size,
                              hipStream_t stream) {
    const float*     x    = (const float*)d_in[0];
    const long long* ei   = (const long long*)d_in[1];   // int64 edge_index [2][N_EDGES]
    const float*     w    = (const float*)d_in[2];       // [512][256]
    const float*     bias = (const float*)d_in[3];       // [256]
    float*           out  = (float*)d_out;               // [10000][256]

    // workspace layout (256B aligned)
    char* ws = (char*)d_ws;
    float*  summed = (float*)(ws);                                   // 10,240,000 B
    float*  deg    = (float*)(ws + 10240000);                        //     40,000 B
    bf16_t* inb    = (bf16_t*)(ws + 10280192);                       // 10,240,000 B
    bf16_t* btw    = (bf16_t*)(ws + 20520192);                       //    262,144 B
    float*  raw    = (float*)(ws + 20782336);                        // 10,240,000 B

    // 1) zero summed + deg (contiguous region)
    {
        unsigned n = (unsigned)N_NODES * D_IN + N_NODES;             // 2,570,000 floats
        zero_kernel<<<(n + 255u) / 256u, 256, 0, stream>>>(summed, n);
    }
    // 2) scatter-add
    {
        unsigned n = (unsigned)N_EDGES * 64u;                        // 20,480,000 threads
        scatter_kernel<<<n / 256u, 256, 0, stream>>>(x, ei, summed, deg);
    }
    // 3) concat + bf16 convert
    {
        unsigned n = (unsigned)N_NODES * D_CAT;                      // 5,120,000
        convert_kernel<<<n / 256u, 256, 0, stream>>>(x, summed, deg, inb);
    }
    // 4) weight transpose + bf16 convert
    {
        unsigned n = (unsigned)D_OUT * D_CAT;                        // 131,072
        wtrans_kernel<<<n / 256u, 256, 0, stream>>>(w, btw);
    }
    // 5) WMMA GEMM: 625*16 = 10000 tiles, 8 waves per 256-thread block
    {
        unsigned tiles = (N_NODES / 16u) * (D_OUT / 16u);            // 10000
        gemm_kernel<<<tiles / 8u, 256, 0, stream>>>(inb, btw, raw);
    }
    // 6) normalize + bias
    finalize_kernel<<<N_NODES, 256, 0, stream>>>(raw, bias, out);
}
